// HumanVAttention_78030965834285
// MI455X (gfx1250) — matrix operand order repaired and checked by reading, not compiled, not run on hardware
//
#include <hip/hip_runtime.h>
#include <cstdint>
#include <cstddef>

// ---------------------------------------------------------------------------
// Problem constants (match reference)
// ---------------------------------------------------------------------------
#define TT    8192
#define HID   2048
#define NH    32
#define NKV   8
#define DH    64
#define BLK   64
#define NB    (TT / BLK)          // 128 query blocks
#define LOCAL 8
#define SCALE 0.125f              // 1/sqrt(64)
#define NEGV  (-1.0e30f)

typedef __attribute__((ext_vector_type(16))) __bf16    bf16x16;
typedef __attribute__((ext_vector_type(8)))  __bf16    bf16x8;
typedef __attribute__((ext_vector_type(8)))  float     f32x8;
typedef __attribute__((ext_vector_type(4)))  uint32_t  u32x4;
typedef __attribute__((ext_vector_type(8)))  int32_t   i32x8;
typedef __attribute__((ext_vector_type(4)))  int32_t   i32x4;

#if defined(__has_builtin)
#if __has_builtin(__builtin_amdgcn_tensor_load_to_lds) && \
    __has_builtin(__builtin_amdgcn_s_wait_tensorcnt)
#define USE_TDM 1
#endif
#endif
#ifndef USE_TDM
#define USE_TDM 0
#endif

// ---------------------------------------------------------------------------
// bf16 <-> f32 helpers (round-to-nearest-even)
// ---------------------------------------------------------------------------
__device__ __forceinline__ __bf16 f2bf(float f) {
  union { float f; uint32_t u; } v; v.f = f;
  uint32_t r = v.u + 0x7FFFu + ((v.u >> 16) & 1u);
  uint16_t h = (uint16_t)(r >> 16);
  return __builtin_bit_cast(__bf16, h);
}
__device__ __forceinline__ float bf2f(__bf16 b) {
  uint16_t h = __builtin_bit_cast(uint16_t, b);
  union { uint32_t u; float f; } v; v.u = ((uint32_t)h) << 16;
  return v.f;
}

// ---------------------------------------------------------------------------
// WMMA fragment loaders (16-bit, wave32; this code's contiguous k order, same for A and B)
// A 16x32: lane<16: row=lane,  elems = K[k0..k0+15]
//          lane>=16: row=lane-16, elems = K[k0+16..k0+31]
// B 32x16: lane = (n&15) + 16*(k>=16); 16 contiguous k per lane
// C/D 16x16 f32: vgpr r -> row = r + 8*(lane>>4); col = lane&15
// ---------------------------------------------------------------------------
__device__ __forceinline__ bf16x16 load_frag_a(const __bf16* __restrict__ base,
                                               int ld, int k0, int lane) {
  const __bf16* p = base + (size_t)(lane & 15) * ld + k0 + ((lane >> 4) << 4);
  return *(const bf16x16*)p;
}
__device__ __forceinline__ bf16x16 load_frag_b(const __bf16* __restrict__ base,
                                               int ld, int k0, int lane) {
  const __bf16* p = base + (size_t)(lane & 15) * ld + k0 + ((lane >> 4) << 4);
  return *(const bf16x16*)p;
}
__device__ __forceinline__ f32x8 wmma_bf16(bf16x16 a, bf16x16 b, f32x8 c) {
  return __builtin_amdgcn_wmma_f32_16x16x32_bf16(false, a, false, b,
                                                 (short)0, c, false, false);
}

// 16-lane (half-wave) reductions via lane shuffles
__device__ __forceinline__ float redmax16(float v) {
#pragma unroll
  for (int off = 8; off >= 1; off >>= 1) {
    float o = __shfl_xor(v, off, 32);
    v = v > o ? v : o;
  }
  return v;
}
__device__ __forceinline__ float redsum16(float v) {
#pragma unroll
  for (int off = 8; off >= 1; off >>= 1)
    v += __shfl_xor(v, off, 32);
  return v;
}

#if USE_TDM
// ---------------------------------------------------------------------------
// Tensor Data Mover: 2D tile load Global -> LDS (CDNA5 ISA).
// D# group0: count=1 user descriptor, lds byte addr, 57-bit global addr,
// type=2 ("image").  group1: data_size=2B, tile WxH, row stride (elements).
// Tensor dims set large so the in-bounds tile never trips OOB handling.
// Issue once per workgroup (TDM ignores EXEC; one instruction = whole tile).
// ---------------------------------------------------------------------------
__device__ __forceinline__ void tdm_load_2d_bf16(uint32_t lds_byte_addr,
                                                 const void* gaddr,
                                                 uint32_t tile_d0,
                                                 uint32_t tile_d1,
                                                 uint32_t stride_d0) {
  uint64_t ga = (uint64_t)(uintptr_t)gaddr;
  u32x4 g0;
  g0[0] = 1u;                                   // count=1, user mode
  g0[1] = lds_byte_addr;                        // LDS destination (bytes)
  g0[2] = (uint32_t)(ga & 0xFFFFFFFFu);         // global addr [31:0]
  g0[3] = (uint32_t)((ga >> 32) & 0x01FFFFFFu)  // global addr [56:32]
          | (2u << 30);                         // type = 2
  const uint32_t td0 = 0x00100000u;             // generous tensor dims
  const uint32_t td1 = 0x00100000u;
  i32x8 g1;
  g1[0] = (int32_t)(1u << 16);                  // data_size = 1 (2 bytes)
  g1[1] = (int32_t)((td0 & 0xFFFFu) << 16);     // tensor_dim0 [15:0]
  g1[2] = (int32_t)(((td0 >> 16) & 0xFFFFu) |   // tensor_dim0 [31:16]
                    ((td1 & 0xFFFFu) << 16));   // tensor_dim1 [15:0]
  g1[3] = (int32_t)(((td1 >> 16) & 0xFFFFu) |   // tensor_dim1 [31:16]
                    ((tile_d0 & 0xFFFFu) << 16)); // tile_dim0
  g1[4] = (int32_t)(tile_d1 & 0xFFFFu);         // tile_dim1 (tile_dim2 = 0)
  g1[5] = (int32_t)stride_d0;                   // tensor_dim0_stride [31:0]
  g1[6] = 0;                                    // stride hi + dim1_stride lo
  g1[7] = 0;
  i32x4 z4 = {0, 0, 0, 0};
#if defined(__clang_major__) && (__clang_major__ >= 23)
  i32x8 z8 = {0, 0, 0, 0, 0, 0, 0, 0};
  __builtin_amdgcn_tensor_load_to_lds(g0, g1, z4, z4, z8, 0);
#else
  __builtin_amdgcn_tensor_load_to_lds(g0, g1, z4, z4, 0);
#endif
}
#endif  // USE_TDM

// ---------------------------------------------------------------------------
// Kernel 1: f32 -> bf16 conversion (vectorized x4; n % 4 == 0)
// ---------------------------------------------------------------------------
__global__ void k_f32_to_bf16(const float* __restrict__ in,
                              __bf16* __restrict__ out, int n) {
  int i = (blockIdx.x * blockDim.x + threadIdx.x) * 4;
  if (i >= n) return;
  float4 v = *(const float4*)(in + i);
  out[i + 0] = f2bf(v.x);
  out[i + 1] = f2bf(v.y);
  out[i + 2] = f2bf(v.z);
  out[i + 3] = f2bf(v.w);
}

// ---------------------------------------------------------------------------
// Kernel 2: C[M,N] = A[M,K] * W[N,K]^T  (bf16 in, f32 accumulate)
// Wave tile 64x64 (16 wmma accumulators), block = 4 waves = 256 rows/block-y.
// Output to f32 (Cf) or bf16 (Cb), whichever is non-null.
// ---------------------------------------------------------------------------
__global__ __launch_bounds__(128)
void k_gemm_bf16(const __bf16* __restrict__ A, const __bf16* __restrict__ W,
                 float* __restrict__ Cf, __bf16* __restrict__ Cb,
                 int M, int N, int K) {
  const int lane = threadIdx.x & 31;
  const int wave = threadIdx.x >> 5;
  const int n0 = blockIdx.x * 64;
  const int m0 = (blockIdx.y * 4 + wave) * 64;
  if (m0 >= M || n0 >= N) return;

  f32x8 acc[4][4] = {};

  for (int k0 = 0; k0 < K; k0 += 32) {
    if (k0 + 32 < K) {  // hint next K-step into cache (global_prefetch_b8)
      __builtin_prefetch(A + (size_t)(m0 + (lane & 15) * 4) * K + k0 + 32, 0, 1);
      __builtin_prefetch(W + (size_t)(n0 + (lane & 15) * 4) * K + k0 + 32, 0, 1);
    }
    bf16x16 a[4], b[4];
#pragma unroll
    for (int i = 0; i < 4; ++i)
      a[i] = load_frag_a(A + (size_t)(m0 + i * 16) * K, K, k0, lane);
#pragma unroll
    for (int j = 0; j < 4; ++j)
      b[j] = load_frag_b(W + (size_t)(n0 + j * 16) * K, K, k0, lane);
#pragma unroll
    for (int i = 0; i < 4; ++i)
#pragma unroll
      for (int j = 0; j < 4; ++j)
        acc[i][j] = wmma_bf16(a[i], b[j], acc[i][j]);
  }

  // Branch hoisted out of the store loops; int offsets off a single base.
  const int nl = lane & 15;
  const int hw = lane >> 4;
  if (Cf) {
    float* cp = Cf + (size_t)(m0 + 8 * hw) * N + (n0 + nl);
#pragma unroll
    for (int i = 0; i < 4; ++i)
#pragma unroll
      for (int r = 0; r < 8; ++r) {
        float* rp = cp + (size_t)(i * 16 + r) * N;
#pragma unroll
        for (int j = 0; j < 4; ++j) rp[j * 16] = acc[i][j][r];
      }
  } else {
    __bf16* cp = Cb + (size_t)(m0 + 8 * hw) * N + (n0 + nl);
#pragma unroll
    for (int i = 0; i < 4; ++i)
#pragma unroll
      for (int r = 0; r < 8; ++r) {
        __bf16* rp = cp + (size_t)(i * 16 + r) * N;
#pragma unroll
        for (int j = 0; j < 4; ++j) rp[j * 16] = f2bf(acc[i][j][r]);
      }
  }
}

// ---------------------------------------------------------------------------
// Kernel 3: RoPE in-place on bf16 [T][heads*D]; one thread per (t, h, pair)
// ---------------------------------------------------------------------------
__global__ void k_rope(__bf16* __restrict__ X, int heads) {
  const int half = DH / 2;  // 32
  int idx = blockIdx.x * blockDim.x + threadIdx.x;
  int p = idx % half;
  int h = (idx / half) % heads;
  int t = idx / (half * heads);
  if (t >= TT) return;
  float inv = __powf(10000.0f, -(float)(2 * p) / (float)DH);
  float ang = (float)t * inv;
  float c, s;
  __sincosf(ang, &s, &c);
  __bf16* base = X + ((size_t)t * heads + h) * DH;
  float x1 = bf2f(base[p]);
  float x2 = bf2f(base[p + half]);
  base[p]        = f2bf(x1 * c - x2 * s);
  base[p + half] = f2bf(x2 * c + x1 * s);
}

// ---------------------------------------------------------------------------
// Kernel 4: block-sparse flash attention.
// grid = (NB query blocks, NH heads); block = 128 threads = 4 waves.
// Wave w handles query rows [i*64 + 16w, +16).  Online softmax over the
// selected key blocks {0} U {max(1,i-7)..i}; only j==i needs causal mask.
// V tile staged to LDS by the Tensor Data Mover (one tensor_load_to_lds per
// key block, TENSORcnt-waited), then transposed LDS->LDS for B-fragments.
// ---------------------------------------------------------------------------
__global__ __launch_bounds__(128)
void k_attn(const __bf16* __restrict__ Q, const __bf16* __restrict__ Kx,
            const __bf16* __restrict__ Vx, __bf16* __restrict__ O) {
  const int i    = blockIdx.x;       // query block
  const int h    = blockIdx.y;       // head
  const int hkv  = h >> 2;           // GQA group of 4
  const int lane = threadIdx.x & 31;
  const int wave = threadIdx.x >> 5;
  const int ldq  = NH * DH;          // 2048
  const int ldk  = NKV * DH;         // 512

  __shared__ __bf16 VL[BLK * DH];        // V tile  [key][d]  8 KB (TDM dest)
  __shared__ __bf16 VT[DH * BLK];        // V^T     [d][key]  8 KB
  __shared__ __bf16 PS[4][16 * BLK];     // P strip per wave  8 KB

  // Q fragments for this wave's 16 rows (kept in registers all kernel)
  const __bf16* qbase = Q + (size_t)(i * BLK + wave * 16) * ldq + h * DH;
  bf16x16 aq0 = load_frag_a(qbase, ldq, 0,  lane);
  bf16x16 aq1 = load_frag_a(qbase, ldq, 32, lane);

  f32x8 o[4] = {};
  float mrow[8], lrow[8];
#pragma unroll
  for (int r = 0; r < 8; ++r) { mrow[r] = NEGV; lrow[r] = 0.0f; }

  int blocks[LOCAL + 1];
  int nblk = 0;
  blocks[nblk++] = 0;
  int blo = (i - (LOCAL - 1) < 1) ? 1 : i - (LOCAL - 1);
  for (int b = blo; b <= i; ++b) blocks[nblk++] = b;

  for (int t = 0; t < nblk; ++t) {
    const int j = blocks[t];

    // ---- stage V tile into LDS ----
    __syncthreads();   // protect VL/VT against previous iteration's readers
#if USE_TDM
    if (wave == 0) {
      tdm_load_2d_bf16((uint32_t)(uintptr_t)(void*)VL,
                       Vx + (size_t)(j * BLK) * ldk + hkv * DH,
                       /*tile_d0=*/DH, /*tile_d1=*/BLK, /*stride=*/ldk);
      __builtin_amdgcn_s_wait_tensorcnt(0);
    }
    __syncthreads();
    // LDS -> LDS transpose: VT[d][key] = VL[key][d]
    for (int e = threadIdx.x; e < BLK * (DH / 8); e += 128) {
      int key = e >> 3;
      int dg  = (e & 7) << 3;
      bf16x8 vv = *(const bf16x8*)(VL + key * DH + dg);
#pragma unroll
      for (int u = 0; u < 8; ++u) VT[(dg + u) * BLK + key] = vv[u];
    }
#else
    for (int e = threadIdx.x; e < BLK * (DH / 8); e += 128) {
      int key = e >> 3;
      int dg  = (e & 7) << 3;
      bf16x8 vv = *(const bf16x8*)(Vx + (size_t)(j * BLK + key) * ldk +
                                   hkv * DH + dg);
#pragma unroll
      for (int u = 0; u < 8; ++u) VT[(dg + u) * BLK + key] = vv[u];
    }
#endif
    __syncthreads();

    // ---- S = Q K^T (row-major K rows ARE the Kᵀ B-fragments) ----
    f32x8 s[4];
#pragma unroll
    for (int jt = 0; jt < 4; ++jt) {
      const __bf16* kb = Kx + (size_t)(j * BLK + jt * 16) * ldk + hkv * DH;
      bf16x16 b0 = load_frag_b(kb, ldk, 0,  lane);
      bf16x16 b1 = load_frag_b(kb, ldk, 32, lane);
      f32x8 c = {};
      c = wmma_bf16(aq0, b0, c);
      c = wmma_bf16(aq1, b1, c);
      s[jt] = c;
    }

    // ---- scale + causal mask (diagonal block only) ----
    const bool diag = (j == i);
#pragma unroll
    for (int jt = 0; jt < 4; ++jt)
#pragma unroll
      for (int r = 0; r < 8; ++r) {
        float v = s[jt][r] * SCALE;
        if (diag) {
          int col = jt * 16 + (lane & 15);
          int row = wave * 16 + r + 8 * (lane >> 4);
          if (col > row) v = NEGV;
        }
        s[jt][r] = v;
      }

    // ---- online softmax stats ----
    float fac[8];
#pragma unroll
    for (int r = 0; r < 8; ++r) {
      float mx = s[0][r];
#pragma unroll
      for (int jt = 1; jt < 4; ++jt) mx = mx > s[jt][r] ? mx : s[jt][r];
      mx = redmax16(mx);
      float mnew = mx > mrow[r] ? mx : mrow[r];
      fac[r] = __expf(mrow[r] - mnew);
      mrow[r] = mnew;
      float ps = 0.0f;
#pragma unroll
      for (int jt = 0; jt < 4; ++jt) {
        float p = __expf(s[jt][r] - mnew);
        s[jt][r] = p;
        ps += p;
      }
      lrow[r] = lrow[r] * fac[r] + redsum16(ps);
    }
#pragma unroll
    for (int dt = 0; dt < 4; ++dt)
#pragma unroll
      for (int r = 0; r < 8; ++r) o[dt][r] *= fac[r];

    // ---- re-fragment P (C layout -> A layout) via per-wave LDS strip ----
    __bf16* ps = PS[wave];
#pragma unroll
    for (int jt = 0; jt < 4; ++jt)
#pragma unroll
      for (int r = 0; r < 8; ++r)
        ps[(r + 8 * (lane >> 4)) * BLK + jt * 16 + (lane & 15)] =
            f2bf(s[jt][r]);
    __syncthreads();

    bf16x16 ap0 = load_frag_a(ps, BLK, 0,  lane);
    bf16x16 ap1 = load_frag_a(ps, BLK, 32, lane);

    // ---- O += P V  (B-frags contiguous from transposed-V LDS) ----
#pragma unroll
    for (int dt = 0; dt < 4; ++dt) {
      bf16x16 bv0 = load_frag_b(VT + dt * 16 * BLK, BLK, 0,  lane);
      bf16x16 bv1 = load_frag_b(VT + dt * 16 * BLK, BLK, 32, lane);
      o[dt] = wmma_bf16(ap0, bv0, o[dt]);
      o[dt] = wmma_bf16(ap1, bv1, o[dt]);
    }
  }

  // ---- epilogue: normalize and store bf16 [T][NH*DH] ----
#pragma unroll
  for (int dt = 0; dt < 4; ++dt)
#pragma unroll
    for (int r = 0; r < 8; ++r) {
      int row = i * BLK + wave * 16 + r + 8 * (lane >> 4);
      int d   = dt * 16 + (lane & 15);
      O[(size_t)row * ldq + h * DH + d] = f2bf(o[dt][r] / lrow[r]);
    }
}

// ---------------------------------------------------------------------------
// Host-side launch
// ---------------------------------------------------------------------------
extern "C" void kernel_launch(void* const* d_in, const int* in_sizes, int n_in,
                              void* d_out, int out_size, void* d_ws,
                              size_t ws_size, hipStream_t stream) {
  const float* hs = (const float*)d_in[0];
  const float* Wq = (const float*)d_in[1];
  const float* Wk = (const float*)d_in[2];
  const float* Wv = (const float*)d_in[3];
  const float* Wo = (const float*)d_in[4];
  float* out = (float*)d_out;

  const size_t n_hs = (size_t)TT * HID;          // 16,777,216
  const size_t n_wq = (size_t)(NH * DH) * HID;   //  4,194,304
  const size_t n_wk = (size_t)(NKV * DH) * HID;  //  1,048,576
  const size_t n_wo = (size_t)(NH * DH) * (NH * DH);
  const size_t n_q  = (size_t)TT * NH * DH;
  const size_t n_kv = (size_t)TT * NKV * DH;

  char* p = (char*)d_ws;
  __bf16* hsb = (__bf16*)p;  p += n_hs * 2;
  __bf16* wqb = (__bf16*)p;  p += n_wq * 2;
  __bf16* wkb = (__bf16*)p;  p += n_wk * 2;
  __bf16* wvb = (__bf16*)p;  p += n_wk * 2;
  __bf16* wob = (__bf16*)p;  p += n_wo * 2;
  __bf16* qb  = (__bf16*)p;  p += n_q  * 2;
  __bf16* kb  = (__bf16*)p;  p += n_kv * 2;
  __bf16* vb  = (__bf16*)p;  p += n_kv * 2;
  __bf16* aob = (__bf16*)p;  p += n_q  * 2;

  // 1) convert inputs to bf16
  k_f32_to_bf16<<<(int)(n_hs / 4 / 256), 256, 0, stream>>>(hs, hsb, (int)n_hs);
  k_f32_to_bf16<<<(int)(n_wq / 4 / 256), 256, 0, stream>>>(Wq, wqb, (int)n_wq);
  k_f32_to_bf16<<<(int)(n_wk / 4 / 256), 256, 0, stream>>>(Wk, wkb, (int)n_wk);
  k_f32_to_bf16<<<(int)(n_wk / 4 / 256), 256, 0, stream>>>(Wv, wvb, (int)n_wk);
  k_f32_to_bf16<<<(int)(n_wo / 4 / 256), 256, 0, stream>>>(Wo, wob, (int)n_wo);

  // 2) projections: q = hs @ Wq^T  (M=8192), k/v = hs @ Wk/Wv^T
  k_gemm_bf16<<<dim3((NH * DH) / 64, TT / 256), 128, 0, stream>>>(
      hsb, wqb, nullptr, qb, TT, NH * DH, HID);
  k_gemm_bf16<<<dim3((NKV * DH) / 64, TT / 256), 128, 0, stream>>>(
      hsb, wkb, nullptr, kb, TT, NKV * DH, HID);
  k_gemm_bf16<<<dim3((NKV * DH) / 64, TT / 256), 128, 0, stream>>>(
      hsb, wvb, nullptr, vb, TT, NKV * DH, HID);

  // 3) RoPE on q and k
  k_rope<<<(int)((size_t)TT * NH * (DH / 2) / 256), 256, 0, stream>>>(qb, NH);
  k_rope<<<(int)((size_t)TT * NKV * (DH / 2) / 256), 256, 0, stream>>>(kb, NKV);

  // 4) block-sparse attention
  k_attn<<<dim3(NB, NH), 128, 0, stream>>>(qb, kb, vb, aob);

  // 5) output projection: out = attn @ Wo^T  (f32 out)
  k_gemm_bf16<<<dim3((NH * DH) / 64, TT / 256), 128, 0, stream>>>(
      aob, wob, out, nullptr, TT, NH * DH, NH * DH);
}
